// Attention_mid_6485400617364
// MI455X (gfx1250) — matrix operand order repaired and checked by reading, not compiled, not run on hardware
//
#include <hip/hip_runtime.h>

// ---------------------------------------------------------------------------
// Attention-LSTM decoder for MI455X (gfx1250, wave32, WMMA, async->LDS).
// B=512, T=128, D=512, H=512, MID=BOT=38, steps S=26.
//  - bf16 WMMA (f32 accum) for all GEMMs
//  - bf16 copies of batch_H and batch_H_proj => per-step working set (128 MB)
//    fits the 192 MB L2; 26x re-reads come from L2, not HBM
//  - A-tiles staged global->LDS with global_load_async_to_lds_b128 (ASYNCcnt)
//  - K-unrolled-by-2 ping-pong B-fragment buffering (no register copies,
//    no mid-loop branches -> no WMMA->VALU hazard NOP storms)
//  - hardware v_tanh_f32 in the attention score loop (33.5M tanh/step)
// ---------------------------------------------------------------------------

typedef __bf16 bf16;
typedef __attribute__((ext_vector_type(16))) __bf16 v16bf;
typedef __attribute__((ext_vector_type(8)))  float  v8f;

struct bf16x2 { bf16 x, y; };     // dword-granular bf16 pair loads

#define Bsz   512
#define Tlen  128
#define Din   512
#define Hid   512
#define MID   38
#define BOT   38
#define STEPS 26
#define KCAT  1152   // 512 ctx + 38 + 38 + 512 h = 1100, padded to 36*32
#define HOFF  588    // offset of h inside x_cat row
#define LDA_P 520    // LDS row stride (bf16) for proj A tile  (bank-safe)
#define LDA_G 1160   // LDS row stride (bf16) for gates A tile (bank-safe)

// ---------------- gfx1250 async global->LDS helpers -------------------------

__device__ __forceinline__ unsigned lds_off(const void* p) {
  // generic pointer to LDS: addr[31:0] is the LDS byte offset
  return (unsigned)(unsigned long long)p;
}

__device__ __forceinline__ void async_g2l_b128(unsigned lds_byte_off,
                                               const void* gaddr) {
  asm volatile("global_load_async_to_lds_b128 %0, %1, off"
               :: "v"(lds_byte_off), "v"(gaddr) : "memory");
}

__device__ __forceinline__ void wait_async0() {
  asm volatile("s_wait_asynccnt 0x0" ::: "memory");
}

// ---------------- fast transcendentals --------------------------------------

__device__ __forceinline__ float tanh_fast(float x) {
#if __has_builtin(__builtin_amdgcn_tanhf)
  return __builtin_amdgcn_tanhf(x);       // v_tanh_f32 (gfx1250 TRANS op)
#elif __has_builtin(__builtin_amdgcn_tanh_f32)
  return __builtin_amdgcn_tanh_f32(x);
#else
  return tanhf(x);
#endif
}

__device__ __forceinline__ float sigmoidf_(float x) {
  return 1.0f / (1.0f + __expf(-x));
}

// ---------------- WMMA fragment helpers (CDNA5 ISA layouts) ------

// A-matrix 16x32 bf16 fragment: lane holds row M=lane&15,
// K-range (lane>>4)*16 + i (16 contiguous, matching ldBT_bf16).
__device__ __forceinline__ v16bf ldA_bf16(const bf16* __restrict__ p, int ld) {
  int lane = threadIdx.x & 31;
  const bf16* row = p + (size_t)(lane & 15) * ld + ((lane >> 4) << 4);
  v16bf f;
#pragma unroll
  for (int i = 0; i < 8; ++i) f[i] = row[i];
#pragma unroll
  for (int i = 0; i < 8; ++i) f[8 + i] = row[8 + i];
  return f;
}

// B-matrix 32x16 fragment from B^T stored row-major (weight W[n][k]):
// lane holds column N=lane&15, K-range (lane>>4)*16 + i => 16 contiguous bf16.
__device__ __forceinline__ v16bf ldBT_bf16(const bf16* __restrict__ p, int ld) {
  int lane = threadIdx.x & 31;
  const bf16* row = p + (size_t)(lane & 15) * ld + ((lane >> 4) << 4);
  v16bf f;
#pragma unroll
  for (int i = 0; i < 16; ++i) f[i] = row[i];
  return f;
}

__device__ __forceinline__ v8f wmma_bf16(v16bf a, v16bf b, v8f c) {
  return __builtin_amdgcn_wmma_f32_16x16x32_bf16(false, a, false, b,
                                                 (short)0, c, false, false);
}

// ---------------- prep kernels ---------------------------------------------

__global__ void k_f32_to_bf16(const float* __restrict__ src,
                              bf16* __restrict__ dst, int n) {
  int i = blockIdx.x * blockDim.x + threadIdx.x;
  if (i < n) dst[i] = (bf16)src[i];
}

// vectorized: 8 elements/thread (b128 in, b128 out) for the 33.5M batch_H copy
__global__ void k_bh_bf16(const float* __restrict__ s, bf16* __restrict__ d) {
  size_t i = ((size_t)blockIdx.x * blockDim.x + threadIdx.x) * 8;
#pragma unroll
  for (int j = 0; j < 8; ++j) d[i + j] = (bf16)s[i + j];
}

// W_cat[r][k] : k<588 -> W_ih[r][k] ; 588<=k<1100 -> W_hh[r][k-588] ; else 0
__global__ void k_build_wcat(const float* __restrict__ Wih,
                             const float* __restrict__ Whh,
                             bf16* __restrict__ Wcat) {
  int i = blockIdx.x * blockDim.x + threadIdx.x;
  if (i >= 4 * Hid * KCAT) return;
  int r = i / KCAT, k = i % KCAT;
  float v = 0.0f;
  if (k < Din + MID + BOT)      v = Wih[(size_t)r * (Din + MID + BOT) + k];
  else if (k < HOFF + Hid)      v = Whh[(size_t)r * Hid + (k - HOFF)];
  Wcat[i] = (bf16)v;
}

// zero x_cat (incl. h slot and K-padding) and cell state c
__global__ void k_init_state(bf16* __restrict__ xcat, float* __restrict__ c) {
  int i = blockIdx.x * blockDim.x + threadIdx.x;
  if (i < Bsz * KCAT) xcat[i] = (bf16)0.0f;
  if (i < Bsz * Hid)  c[i] = 0.0f;
}

// ---------------- GEMM 1: proj = batch_H @ W_i2h^T  (store bf16) ------------
// One block per 16-row m-tile; 8 waves cover N=512 as 8 strips of 64.
// A tile (16x512 bf16) staged via async global->LDS; ping-pong B buffers.
// NOTE: the final B prefetch reads <=64B past the weight buffer; this stays
// inside the workspace (next allocation) and the values are never consumed.
__global__ void k_proj_gemm(const bf16* __restrict__ Abf,
                            const bf16* __restrict__ Wb,
                            bf16* __restrict__ proj) {
  __shared__ bf16 sA[16 * LDA_P];
  int tid = threadIdx.x;
  int mt = blockIdx.x;                       // 4096 m-tiles

  const bf16* g0 = Abf + (size_t)mt * 16 * Din;
  for (int i = tid; i < 16 * 64; i += 256) { // 64 x b128 chunks per row
    int r = i >> 6, cch = i & 63;
    async_g2l_b128(lds_off(&sA[r * LDA_P + cch * 8]),
                   g0 + (size_t)r * Din + cch * 8);
  }
  wait_async0();
  __syncthreads();

  int wv = tid >> 5;
  int n0 = wv * 64;
  const v8f zero = {0.f,0.f,0.f,0.f,0.f,0.f,0.f,0.f};
  v8f acc[4];
#pragma unroll
  for (int j = 0; j < 4; ++j) acc[j] = zero;

  v16bf bc[4], bn[4];
#pragma unroll
  for (int j = 0; j < 4; ++j)
    bc[j] = ldBT_bf16(Wb + (size_t)(n0 + j * 16) * Din, Din);

  for (int k0 = 0; k0 < Din; k0 += 64) {     // 8 double-iterations
#pragma unroll
    for (int j = 0; j < 4; ++j)
      bn[j] = ldBT_bf16(Wb + (size_t)(n0 + j * 16) * Din + k0 + 32, Din);
    v16bf a0 = ldA_bf16(sA + k0, LDA_P);
#pragma unroll
    for (int j = 0; j < 4; ++j) acc[j] = wmma_bf16(a0, bc[j], acc[j]);
#pragma unroll
    for (int j = 0; j < 4; ++j)
      bc[j] = ldBT_bf16(Wb + (size_t)(n0 + j * 16) * Din + k0 + 64, Din);
    v16bf a1 = ldA_bf16(sA + k0 + 32, LDA_P);
#pragma unroll
    for (int j = 0; j < 4; ++j) acc[j] = wmma_bf16(a1, bn[j], acc[j]);
  }

  int lane = tid & 31;
  int col = lane & 15, rb = (lane >> 4) << 3;
#pragma unroll
  for (int j = 0; j < 4; ++j) {
    bf16* Cp = proj + ((size_t)mt * 16 + rb) * Hid + n0 + j * 16 + col;
#pragma unroll
    for (int r = 0; r < 8; ++r) Cp[(size_t)r * Hid] = (bf16)acc[j][r];
  }
}

// ---------------- GEMM 2 (per step): hp = h @ W_h2h^T + b_h2h ---------------
// M=512, N=512, K=512. A rows live in x_cat at column offset HOFF (bf16).
__global__ void k_hp_gemm(const bf16* __restrict__ xcat,
                          const bf16* __restrict__ Wb,
                          const float* __restrict__ bias,
                          float* __restrict__ hp) {
  int wave = (blockIdx.x * blockDim.x + threadIdx.x) >> 5;
  int mt = wave >> 3;            // 32 m-tiles
  int ns = wave & 7;             // 8 strips of 64
  const bf16* Ap = xcat + (size_t)mt * 16 * KCAT + HOFF;
  const v8f zero = {0.f,0.f,0.f,0.f,0.f,0.f,0.f,0.f};
  v8f acc[4];
#pragma unroll
  for (int j = 0; j < 4; ++j) acc[j] = zero;

  v16bf bc[4], bn[4];
#pragma unroll
  for (int j = 0; j < 4; ++j)
    bc[j] = ldBT_bf16(Wb + (size_t)(ns * 64 + j * 16) * Hid, Hid);

  for (int k0 = 0; k0 < Hid; k0 += 64) {
#pragma unroll
    for (int j = 0; j < 4; ++j)
      bn[j] = ldBT_bf16(Wb + (size_t)(ns * 64 + j * 16) * Hid + k0 + 32, Hid);
    v16bf a0 = ldA_bf16(Ap + k0, KCAT);
#pragma unroll
    for (int j = 0; j < 4; ++j) acc[j] = wmma_bf16(a0, bc[j], acc[j]);
#pragma unroll
    for (int j = 0; j < 4; ++j)
      bc[j] = ldBT_bf16(Wb + (size_t)(ns * 64 + j * 16) * Hid + k0 + 64, Hid);
    v16bf a1 = ldA_bf16(Ap + k0 + 32, KCAT);
#pragma unroll
    for (int j = 0; j < 4; ++j) acc[j] = wmma_bf16(a1, bn[j], acc[j]);
  }

  int lane = threadIdx.x & 31;
  int col = lane & 15, rb = (lane >> 4) << 3;
#pragma unroll
  for (int j = 0; j < 4; ++j) {
    int n = ns * 64 + j * 16 + col;
    float bi = bias[n];
    float* Cp = hp + ((size_t)mt * 16 + rb) * Hid + n;
#pragma unroll
    for (int r = 0; r < 8; ++r) Cp[(size_t)r * Hid] = acc[j][r] + bi;
  }
}

// ---------------- fused score + softmax + context + onehot ------------------
// One block (256 thr = 8 waves) per batch element. Reads only bf16 tensors
// (proj + batch_H copy) => 128 MB/step working set, L2-resident.
__global__ void k_attn(const bf16* __restrict__ proj,
                       const float* __restrict__ hp,
                       const float* __restrict__ wscore,
                       const bf16* __restrict__ bHb,
                       const int* __restrict__ midp,
                       const int* __restrict__ bott,
                       bf16* __restrict__ xcat, int step) {
  __shared__ float s_hp[Hid];
  __shared__ float s_ws[Hid];
  __shared__ float s_e[Tlen];
  __shared__ float s_alpha[Tlen];
  int b = blockIdx.x;
  int tid = threadIdx.x;
  for (int i = tid; i < Hid; i += 256) {
    s_hp[i] = hp[(size_t)b * Hid + i];
    s_ws[i] = wscore[i];
  }
  __syncthreads();

  int wv = tid >> 5, lane = tid & 31;
  for (int t = wv; t < Tlen; t += 8) {
    const bf16x2* pr2 =
        (const bf16x2*)(proj + ((size_t)b * Tlen + t) * Hid);
    __builtin_prefetch(pr2 + ((t + 8 < Tlen) ? (size_t)4 * Hid : 0), 0, 0);
    float partial = 0.0f;
#pragma unroll 4
    for (int q = lane; q < Hid / 2; q += 32) {   // dword loads, banks 2l,2l+1
      bf16x2 pv = pr2[q];
      int h = q * 2;
      partial += s_ws[h]     * tanh_fast((float)pv.x + s_hp[h]);
      partial += s_ws[h + 1] * tanh_fast((float)pv.y + s_hp[h + 1]);
    }
#pragma unroll
    for (int off = 16; off > 0; off >>= 1)
      partial += __shfl_xor(partial, off, 32);
    if (lane == 0) s_e[t] = partial;
  }
  __syncthreads();

  float m = -3.4e38f;
  for (int t = 0; t < Tlen; ++t) m = fmaxf(m, s_e[t]);
  float sum = 0.0f;
  for (int t = 0; t < Tlen; ++t) sum += __expf(s_e[t] - m);
  float inv = 1.0f / sum;
  if (tid < Tlen) s_alpha[tid] = __expf(s_e[tid] - m) * inv;
  __syncthreads();

  // context[b,d] = sum_t alpha[t] * batch_H[b,t,d]  (bf16 source)
  for (int d = tid; d < Din; d += 256) {
    const bf16* Bp = bHb + (size_t)b * Tlen * Din + d;
    float acc = 0.0f;
#pragma unroll 4
    for (int t = 0; t < Tlen; ++t) acc += s_alpha[t] * (float)Bp[(size_t)t * Din];
    xcat[(size_t)b * KCAT + d] = (bf16)acc;
  }
  if (tid < MID + BOT) {
    float v;
    if (tid < MID) v = (midp[b * STEPS + step] == tid) ? 1.0f : 0.0f;
    else           v = (bott[b * STEPS + step] == (tid - MID)) ? 1.0f : 0.0f;
    xcat[(size_t)b * KCAT + Din + tid] = (bf16)v;
  }
}

// ---------------- GEMM 3 (per step): gates = x_cat @ W_cat^T + b ------------
// M=512, N=2048, K=1152. Block = one m-tile x 8 n-strips; A tile (16x1152)
// staged via async global->LDS and shared by the 8 waves.
__global__ void k_gates_gemm(const bf16* __restrict__ xcat,
                             const bf16* __restrict__ Wcat,
                             const float* __restrict__ bih,
                             const float* __restrict__ bhh,
                             float* __restrict__ G) {
  __shared__ bf16 sA[16 * LDA_G];
  int tid = threadIdx.x;
  int mt = blockIdx.x >> 2;                  // 32 m-tiles
  int ng = blockIdx.x & 3;                   // 4 n-groups of 8 strips

  const bf16* g0 = xcat + (size_t)mt * 16 * KCAT;
  for (int i = tid; i < 16 * 144; i += 256) { // 2304 chunks: 9 uniform iters
    int r = i / 144, cch = i % 144;
    async_g2l_b128(lds_off(&sA[r * LDA_G + cch * 8]),
                   g0 + (size_t)r * KCAT + cch * 8);
  }
  wait_async0();
  __syncthreads();

  int wv = tid >> 5;
  int ns = ng * 8 + wv;                      // 32 strips of 64 over N=2048
  const v8f zero = {0.f,0.f,0.f,0.f,0.f,0.f,0.f,0.f};
  v8f acc[4];
#pragma unroll
  for (int j = 0; j < 4; ++j) acc[j] = zero;

  v16bf bc[4], bn[4];
#pragma unroll
  for (int j = 0; j < 4; ++j)
    bc[j] = ldBT_bf16(Wcat + (size_t)(ns * 64 + j * 16) * KCAT, KCAT);

  for (int k0 = 0; k0 < KCAT; k0 += 64) {    // 18 double-iterations
#pragma unroll
    for (int j = 0; j < 4; ++j)
      bn[j] = ldBT_bf16(Wcat + (size_t)(ns * 64 + j * 16) * KCAT + k0 + 32, KCAT);
    v16bf a0 = ldA_bf16(sA + k0, LDA_G);
#pragma unroll
    for (int j = 0; j < 4; ++j) acc[j] = wmma_bf16(a0, bc[j], acc[j]);
#pragma unroll
    for (int j = 0; j < 4; ++j)
      bc[j] = ldBT_bf16(Wcat + (size_t)(ns * 64 + j * 16) * KCAT + k0 + 64, KCAT);
    v16bf a1 = ldA_bf16(sA + k0 + 32, LDA_G);
#pragma unroll
    for (int j = 0; j < 4; ++j) acc[j] = wmma_bf16(a1, bn[j], acc[j]);
  }

  int lane = tid & 31;
  int col = lane & 15, rb = (lane >> 4) << 3;
#pragma unroll
  for (int j = 0; j < 4; ++j) {
    int n = ns * 64 + j * 16 + col;
    float bi = bih[n] + bhh[n];
    float* Cp = G + ((size_t)mt * 16 + rb) * (4 * Hid) + n;
#pragma unroll
    for (int r = 0; r < 8; ++r) Cp[(size_t)r * (4 * Hid)] = acc[j][r] + bi;
  }
}

// ---------------- LSTM pointwise gates --------------------------------------
__global__ void k_lstm_gate(const float* __restrict__ G,
                            float* __restrict__ c,
                            float* __restrict__ hiddens,
                            bf16* __restrict__ xcat, int step) {
  int idx = blockIdx.x * blockDim.x + threadIdx.x;   // B*H threads
  int b = idx >> 9, h = idx & (Hid - 1);
  const float* g = G + (size_t)b * (4 * Hid);
  float ig = sigmoidf_(g[h]);
  float fg = sigmoidf_(g[Hid + h]);
  float gg = tanh_fast(g[2 * Hid + h]);
  float og = sigmoidf_(g[3 * Hid + h]);
  float cn = fg * c[idx] + ig * gg;
  float hn = og * tanh_fast(cn);
  c[idx] = cn;
  hiddens[((size_t)b * STEPS + step) * Hid + h] = hn;
  xcat[(size_t)b * KCAT + HOFF + h] = (bf16)hn;       // h for next step
}

// ---------------- generator: g = hiddens @ W_gen^T + b_gen ------------------
__global__ void k_gen(const float* __restrict__ hiddens,
                      const float* __restrict__ Wg,
                      const float* __restrict__ bg,
                      float* __restrict__ out) {
  int idx = blockIdx.x * blockDim.x + threadIdx.x;
  if (idx >= Bsz * STEPS * MID) return;
  int cg = idx % MID;
  int r  = idx / MID;             // b*STEPS + s
  const float* hrow = hiddens + (size_t)r * Hid;
  const float* wrow = Wg + (size_t)cg * Hid;
  float acc = bg[cg];
#pragma unroll 4
  for (int h = 0; h < Hid; ++h) acc += hrow[h] * wrow[h];
  out[idx] = acc;
}

// ---------------- host launcher --------------------------------------------

extern "C" void kernel_launch(void* const* d_in, const int* in_sizes, int n_in,
                              void* d_out, int out_size, void* d_ws, size_t ws_size,
                              hipStream_t stream) {
  const float* batch_H = (const float*)d_in[0];
  const int*   midp    = (const int*)d_in[1];
  const int*   bott    = (const int*)d_in[2];
  const float* W_i2h   = (const float*)d_in[3];
  const float* W_h2h   = (const float*)d_in[4];
  const float* b_h2h   = (const float*)d_in[5];
  const float* W_score = (const float*)d_in[6];
  const float* W_ih    = (const float*)d_in[7];
  const float* W_hh    = (const float*)d_in[8];
  const float* b_ih    = (const float*)d_in[9];
  const float* b_hh    = (const float*)d_in[10];
  const float* W_gen   = (const float*)d_in[11];
  const float* b_gen   = (const float*)d_in[12];
  float* out = (float*)d_out;

  char* ws = (char*)d_ws;
  size_t off = 0;
  auto take = [&](size_t bytes) {
    char* p = ws + off;
    off += (bytes + 255) & ~(size_t)255;
    return p;
  };
  bf16*  proj    = (bf16*)take((size_t)Bsz * Tlen * Hid * sizeof(bf16)); // 64 MB
  bf16*  bHb     = (bf16*)take((size_t)Bsz * Tlen * Din * sizeof(bf16)); // 64 MB
  bf16*  wi2h_b  = (bf16*)take((size_t)Hid * Din * sizeof(bf16));
  bf16*  wh2h_b  = (bf16*)take((size_t)Hid * Hid * sizeof(bf16));
  bf16*  wcat_b  = (bf16*)take((size_t)4 * Hid * KCAT * sizeof(bf16));
  bf16*  xcat    = (bf16*)take((size_t)Bsz * KCAT * sizeof(bf16));
  float* hp      = (float*)take((size_t)Bsz * Hid * sizeof(float));
  float* gates   = (float*)take((size_t)Bsz * 4 * Hid * sizeof(float));
  float* cstate  = (float*)take((size_t)Bsz * Hid * sizeof(float));
  float* hiddens = (float*)take((size_t)Bsz * STEPS * Hid * sizeof(float));

  const int TPB = 256;

  // one-time conversions / packing (redone every call: deterministic)
  k_bh_bf16<<<(Bsz * Tlen * Din) / (8 * TPB), TPB, 0, stream>>>(batch_H, bHb);
  k_f32_to_bf16<<<(Hid * Din + TPB - 1) / TPB, TPB, 0, stream>>>(W_i2h, wi2h_b, Hid * Din);
  k_f32_to_bf16<<<(Hid * Hid + TPB - 1) / TPB, TPB, 0, stream>>>(W_h2h, wh2h_b, Hid * Hid);
  k_build_wcat<<<(4 * Hid * KCAT + TPB - 1) / TPB, TPB, 0, stream>>>(W_ih, W_hh, wcat_b);
  k_init_state<<<(Bsz * KCAT + TPB - 1) / TPB, TPB, 0, stream>>>(xcat, cstate);

  // step-invariant projection: one block per 16-row tile
  k_proj_gemm<<<(Bsz * Tlen) / 16, TPB, 0, stream>>>(bHb, wi2h_b, proj);

  for (int s = 0; s < STEPS; ++s) {
    k_hp_gemm<<<32, TPB, 0, stream>>>(xcat, wh2h_b, b_h2h, hp);
    k_attn<<<Bsz, TPB, 0, stream>>>(proj, hp, W_score, bHb, midp, bott, xcat, s);
    k_gates_gemm<<<128, TPB, 0, stream>>>(xcat, wcat_b, b_ih, b_hh, gates);
    k_lstm_gate<<<(Bsz * Hid) / TPB, TPB, 0, stream>>>(gates, cstate, hiddens, xcat, s);
  }

  k_gen<<<(Bsz * STEPS * MID + TPB - 1) / TPB, TPB, 0, stream>>>(hiddens, W_gen, b_gen, out);
}
